// WLNEncoder_12876311954003
// MI455X (gfx1250) — matrix operand order repaired and checked by reading, not compiled, not run on hardware
//
#include <hip/hip_runtime.h>
#include <hip/hip_bf16.h>

// ---------------------------------------------------------------------------
// WLN graph encoder for MI455X (gfx1250, wave32, WMMA).
//
// Precision strategy: bf16 storage / WMMA bf16 inputs, f32 accumulation.
//  - h_atom (25.6MB bf16) + h_bond (51.2MB bf16) fit in 192MB L2 -> the
//    random neighbor gathers are L2 hits, not HBM.
//  - 107 GFLOP of 64x64-K GEMMs run on v_wmma_f32_16x16x32_bf16.
//
// Dead-code pruning vs reference loop:
//  - depths 0,1: only the V/U1/U2 update path (c_atom is overwritten).
//  - depth 2: only the W0/W1/W2 c_atom path (h update never read).
// ---------------------------------------------------------------------------

#define N_ATOMS 200000
#define N_BONDS 400000
#define MAX_NEI 10
#define N_MOLS  8000
#define HS      64

typedef __attribute__((ext_vector_type(16))) __bf16 v16bf;
typedef __attribute__((ext_vector_type(8)))  float  v8f;

union BfTile { v16bf v; uint4 q[2]; };

// A operand (16x32 bf16), hardware layout: lane l holds row m=l&15;
// elements 0..7  = K k0+sel*8 .. +7, elements 8..15 = K k0+16+sel*8 .. +7.
static __device__ inline v16bf load_a_row(const __bf16* rowp, int k0, int sel) {
    BfTile t;
    t.q[0] = *(const uint4*)(rowp + k0 + sel * 8);
    t.q[1] = *(const uint4*)(rowp + k0 + 16 + sel * 8);
    return t.v;
}

// B operand (32x16 bf16) from a [N][Ktot] transposed weight: lane l holds
// column n=l&15; elements 0..7 = K kt*32+sel*8 .. +7, elements 8..15 = K kt*32+16+sel*8 .. +7.
static __device__ inline v16bf load_b_w(const __bf16* wT, int Ktot, int nt, int kt, int lane) {
    const int n  = nt * 16 + (lane & 15);
    const int kh = (lane >> 4) * 8;
    const __bf16* p = wT + n * Ktot + kt * 32 + kh;
    BfTile t;
    t.q[0] = *(const uint4*)(p);
    t.q[1] = *(const uint4*)(p + 16);
    return t.v;
}

static __device__ inline v8f wmma_bf16(v16bf a, v16bf b, v8f c) {
    return __builtin_amdgcn_wmma_f32_16x16x32_bf16(false, a, false, b, (short)0, c, false, false);
}

// ---------------------------------------------------------------------------
// Prep: transpose f32 [K,N] weight -> bf16 [N,K]
// ---------------------------------------------------------------------------
__global__ void transpose_cvt(const float* __restrict__ in, __bf16* __restrict__ out,
                              int K, int N) {
    int i = blockIdx.x * blockDim.x + threadIdx.x;
    if (i >= K * N) return;
    int n = i / K, k = i - n * K;
    out[i] = (__bf16)in[k * N + n];   // out[n*K + k]
}

__global__ void zero_f32(float* __restrict__ p, int n) {
    int i = blockIdx.x * blockDim.x + threadIdx.x;
    if (i < n) p[i] = 0.0f;
}

// ---------------------------------------------------------------------------
// Embeddings (tiny FLOP count -> simple VALU, bf16 output)
// ---------------------------------------------------------------------------
__global__ __launch_bounds__(256) void embed_atom(const float* __restrict__ feat,
                                                  const float* __restrict__ W,  // [87,64]
                                                  __bf16* __restrict__ h) {
    int t = blockIdx.x * 256 + threadIdx.x;
    int n = t & (HS - 1);
    int a = t >> 6;
    if (a >= N_ATOMS) return;
    const float* f = feat + (size_t)a * 87;
    float acc = 0.0f;
    for (int k = 0; k < 87; ++k) acc = fmaf(f[k], W[k * HS + n], acc);
    h[(size_t)a * HS + n] = (__bf16)acc;
}

__global__ __launch_bounds__(256) void embed_bond(const float* __restrict__ feat,
                                                  const float* __restrict__ W,  // [6,64]
                                                  __bf16* __restrict__ h) {
    int t = blockIdx.x * 256 + threadIdx.x;
    int n = t & (HS - 1);
    int e = t >> 6;
    if (e >= N_BONDS) return;
    const float* f = feat + (size_t)e * 6;
    float acc = 0.0f;
#pragma unroll
    for (int k = 0; k < 6; ++k) acc = fmaf(f[k], W[k * HS + n], acc);
    h[(size_t)e * HS + n] = (__bf16)acc;
}

// ---------------------------------------------------------------------------
// Update pass (depths 0,1):
//   nei_label = sum_k relu([h_bond_nei | h_atom_nei] @ V)
//   h_out     = relu(h_in @ U1 + nei_label @ U2)
// 128 threads = 4 waves; each wave owns 16 atoms. 176 WMMA per wave.
// ---------------------------------------------------------------------------
__global__ __launch_bounds__(128) void wln_update(
    const __bf16* __restrict__ h_in, const __bf16* __restrict__ h_bond,
    const int* __restrict__ atom_graph, const int* __restrict__ bond_graph,
    const __bf16* __restrict__ VT,   // [64][128]
    const __bf16* __restrict__ U1T,  // [64][64]
    const __bf16* __restrict__ U2T,  // [64][64]
    __bf16* __restrict__ h_out) {
    __shared__ __attribute__((aligned(16))) __bf16 sV [64 * 128];
    __shared__ __attribute__((aligned(16))) __bf16 sU1[64 * 64];
    __shared__ __attribute__((aligned(16))) __bf16 sU2[64 * 64];
    __shared__ __attribute__((aligned(16))) __bf16 sLbl[4][16 * HS];

    const int t = threadIdx.x;
    for (int i = t; i < (64 * 128) / 8; i += 128) ((uint4*)sV)[i]  = ((const uint4*)VT)[i];
    for (int i = t; i < (64 * 64) / 8;  i += 128) ((uint4*)sU1)[i] = ((const uint4*)U1T)[i];
    for (int i = t; i < (64 * 64) / 8;  i += 128) ((uint4*)sU2)[i] = ((const uint4*)U2T)[i];
    __syncthreads();

    const int wave = t >> 5, lane = t & 31;
    const int m = lane & 15, sel = lane >> 4;
    const int atom0 = (blockIdx.x * 4 + wave) * 16;
    const int arow  = atom0 + m;             // this lane's atom row

    v8f lbl[4] = {};                          // nei_label f32 accum, 4 N-tiles

    const int* ag = atom_graph + arow * MAX_NEI;
    const int* bg = bond_graph + arow * MAX_NEI;
    for (int k = 0; k < MAX_NEI; ++k) {
        const __bf16* brow = h_bond + (size_t)bg[k] * HS;  // gather
        const __bf16* nrow = h_in   + (size_t)ag[k] * HS;  // gather
        v16bf a0 = load_a_row(brow, 0, sel);   // concat K 0..31   (bond)
        v16bf a1 = load_a_row(brow, 32, sel);  // concat K 32..63  (bond)
        v16bf a2 = load_a_row(nrow, 0, sel);   // concat K 64..95  (atom)
        v16bf a3 = load_a_row(nrow, 32, sel);  // concat K 96..127 (atom)
#pragma unroll
        for (int nt = 0; nt < 4; ++nt) {
            v8f acc = {};
            acc = wmma_bf16(a0, load_b_w(sV, 128, nt, 0, lane), acc);
            acc = wmma_bf16(a1, load_b_w(sV, 128, nt, 1, lane), acc);
            acc = wmma_bf16(a2, load_b_w(sV, 128, nt, 2, lane), acc);
            acc = wmma_bf16(a3, load_b_w(sV, 128, nt, 3, lane), acc);
#pragma unroll
            for (int i = 0; i < 8; ++i) lbl[nt][i] += fmaxf(acc[i], 0.0f);  // relu+sum_k
        }
    }

    // Repack nei_label (C layout) -> row-major bf16 in LDS so it can re-enter
    // as an A operand of the @U2 matmul.
    __bf16* lb = sLbl[wave];
#pragma unroll
    for (int nt = 0; nt < 4; ++nt)
#pragma unroll
        for (int r = 0; r < 8; ++r)
            lb[(sel * 8 + r) * HS + nt * 16 + m] = (__bf16)lbl[nt][r];
    __syncthreads();

    const __bf16* hrow = h_in + (size_t)arow * HS;
    v16bf ha0 = load_a_row(hrow, 0, sel);
    v16bf ha1 = load_a_row(hrow, 32, sel);
    const __bf16* lrow = lb + m * HS;
    v16bf la0 = load_a_row(lrow, 0, sel);
    v16bf la1 = load_a_row(lrow, 32, sel);

#pragma unroll
    for (int nt = 0; nt < 4; ++nt) {
        v8f acc = {};
        acc = wmma_bf16(ha0, load_b_w(sU1, 64, nt, 0, lane), acc);
        acc = wmma_bf16(ha1, load_b_w(sU1, 64, nt, 1, lane), acc);
        acc = wmma_bf16(la0, load_b_w(sU2, 64, nt, 0, lane), acc);
        acc = wmma_bf16(la1, load_b_w(sU2, 64, nt, 1, lane), acc);
#pragma unroll
        for (int r = 0; r < 8; ++r)  // C layout: (M=sel*8+r, N=nt*16+m); lanes 0-15 coalesce per row
            h_out[(size_t)(atom0 + sel * 8 + r) * HS + nt * 16 + m] = (__bf16)fmaxf(acc[r], 0.0f);
    }
}

// ---------------------------------------------------------------------------
// Final pass (depth 2):
//   c_atom = (h @ W0) * sum_k (h_atom_nei @ W2) * (h_bond_nei @ W1)
//   c_mol  = segment_sum(c_atom, mol_ids)   [atomic f32 adds]
// 168 WMMA per wave.
// ---------------------------------------------------------------------------
__global__ __launch_bounds__(128) void wln_final(
    const __bf16* __restrict__ h_in, const __bf16* __restrict__ h_bond,
    const int* __restrict__ atom_graph, const int* __restrict__ bond_graph,
    const int* __restrict__ mol_ids,
    const __bf16* __restrict__ W0T, const __bf16* __restrict__ W1T,
    const __bf16* __restrict__ W2T,
    float* __restrict__ c_mol, float* __restrict__ c_atom) {
    __shared__ __attribute__((aligned(16))) __bf16 sW0[64 * 64];
    __shared__ __attribute__((aligned(16))) __bf16 sW1[64 * 64];
    __shared__ __attribute__((aligned(16))) __bf16 sW2[64 * 64];

    const int t = threadIdx.x;
    for (int i = t; i < (64 * 64) / 8; i += 128) {
        ((uint4*)sW0)[i] = ((const uint4*)W0T)[i];
        ((uint4*)sW1)[i] = ((const uint4*)W1T)[i];
        ((uint4*)sW2)[i] = ((const uint4*)W2T)[i];
    }
    __syncthreads();

    const int wave = t >> 5, lane = t & 31;
    const int m = lane & 15, sel = lane >> 4;
    const int atom0 = (blockIdx.x * 4 + wave) * 16;
    const int arow  = atom0 + m;

    const __bf16* hrow = h_in + (size_t)arow * HS;
    v16bf ha0 = load_a_row(hrow, 0, sel);
    v16bf ha1 = load_a_row(hrow, 32, sel);
    v8f fa[4];
#pragma unroll
    for (int nt = 0; nt < 4; ++nt) {
        v8f acc = {};
        acc = wmma_bf16(ha0, load_b_w(sW0, 64, nt, 0, lane), acc);
        acc = wmma_bf16(ha1, load_b_w(sW0, 64, nt, 1, lane), acc);
        fa[nt] = acc;
    }

    v8f s[4] = {};
    const int* ag = atom_graph + arow * MAX_NEI;
    const int* bg = bond_graph + arow * MAX_NEI;
    for (int k = 0; k < MAX_NEI; ++k) {
        const __bf16* brow = h_bond + (size_t)bg[k] * HS;
        const __bf16* nrow = h_in   + (size_t)ag[k] * HS;
        v16bf b0 = load_a_row(brow, 0, sel), b1 = load_a_row(brow, 32, sel);
        v16bf a0 = load_a_row(nrow, 0, sel), a1 = load_a_row(nrow, 32, sel);
#pragma unroll
        for (int nt = 0; nt < 4; ++nt) {
            v8f fb = {};
            fb = wmma_bf16(b0, load_b_w(sW1, 64, nt, 0, lane), fb);
            fb = wmma_bf16(b1, load_b_w(sW1, 64, nt, 1, lane), fb);
            v8f fn = {};
            fn = wmma_bf16(a0, load_b_w(sW2, 64, nt, 0, lane), fn);
            fn = wmma_bf16(a1, load_b_w(sW2, 64, nt, 1, lane), fn);
#pragma unroll
            for (int i = 0; i < 8; ++i) s[nt][i] += fb[i] * fn[i];  // same C layout -> lane-local
        }
    }

    int mymol = mol_ids[atom0 + m];  // lanes 0..15 hold rows 0..15 of the tile
#pragma unroll
    for (int nt = 0; nt < 4; ++nt) {
#pragma unroll
        for (int r = 0; r < 8; ++r) {
            const int mm = sel * 8 + r;
            const int nn = nt * 16 + m;
            const float c = fa[nt][r] * s[nt][r];
            c_atom[(size_t)(atom0 + mm) * HS + nn] = c;
            const int mol = __shfl(mymol, mm, 32);
            atomicAdd(c_mol + (size_t)mol * HS + nn, c);
        }
    }
}

// ---------------------------------------------------------------------------
// Host-side orchestration
// ---------------------------------------------------------------------------
extern "C" void kernel_launch(void* const* d_in, const int* in_sizes, int n_in,
                              void* d_out, int out_size, void* d_ws, size_t ws_size,
                              hipStream_t stream) {
    const float* atom_feat   = (const float*)d_in[0];
    const float* bond_feat   = (const float*)d_in[1];
    const int*   atom_graph  = (const int*)d_in[2];
    const int*   bond_graph  = (const int*)d_in[3];
    const int*   mol_ids     = (const int*)d_in[4];
    const float* W_atom_emb  = (const float*)d_in[5];
    const float* W_bond_emb  = (const float*)d_in[6];
    const float* U1 = (const float*)d_in[7];
    const float* U2 = (const float*)d_in[8];
    const float* V  = (const float*)d_in[9];
    const float* W0 = (const float*)d_in[10];
    const float* W1 = (const float*)d_in[11];
    const float* W2 = (const float*)d_in[12];

    // Workspace layout (bf16), all 16B aligned:
    char* ws = (char*)d_ws;
    __bf16* hA0 = (__bf16*)(ws);                       // 200000*64*2 = 25.6 MB
    __bf16* hA1 = (__bf16*)(ws + 25600000);            // 25.6 MB
    __bf16* hB  = (__bf16*)(ws + 51200000);            // 400000*64*2 = 51.2 MB
    __bf16* W0T = (__bf16*)(ws + 102400000);           // transposed bf16 weights
    __bf16* W1T = W0T + 64 * 64;
    __bf16* W2T = W1T + 64 * 64;
    __bf16* U1T = W2T + 64 * 64;
    __bf16* U2T = U1T + 64 * 64;
    __bf16* VT  = U2T + 64 * 64;                       // [64][128]

    // 1) weight prep: [K,N] f32 -> [N,K] bf16
    transpose_cvt<<<(64 * 64 + 255) / 256, 256, 0, stream>>>(W0, W0T, 64, 64);
    transpose_cvt<<<(64 * 64 + 255) / 256, 256, 0, stream>>>(W1, W1T, 64, 64);
    transpose_cvt<<<(64 * 64 + 255) / 256, 256, 0, stream>>>(W2, W2T, 64, 64);
    transpose_cvt<<<(64 * 64 + 255) / 256, 256, 0, stream>>>(U1, U1T, 64, 64);
    transpose_cvt<<<(64 * 64 + 255) / 256, 256, 0, stream>>>(U2, U2T, 64, 64);
    transpose_cvt<<<(128 * 64 + 255) / 256, 256, 0, stream>>>(V, VT, 128, 64);

    // 2) embeddings
    embed_atom<<<(N_ATOMS * HS) / 256, 256, 0, stream>>>(atom_feat, W_atom_emb, hA0);
    embed_bond<<<(N_BONDS * HS) / 256, 256, 0, stream>>>(bond_feat, W_bond_emb, hB);

    // 3) depths 0,1: ping-pong h buffers
    const int blocks = N_ATOMS / 64;  // 3125, 4 waves x 16 atoms each, exact
    wln_update<<<blocks, 128, 0, stream>>>(hA0, hB, atom_graph, bond_graph, VT, U1T, U2T, hA1);
    wln_update<<<blocks, 128, 0, stream>>>(hA1, hB, atom_graph, bond_graph, VT, U1T, U2T, hA0);

    // 4) depth 2: c_atom + segment-sum into c_mol (d_out is poisoned -> zero first)
    float* c_mol  = (float*)d_out;                 // [8000,64]
    float* c_atom = (float*)d_out + N_MOLS * HS;   // [200000,64]
    zero_f32<<<(N_MOLS * HS + 255) / 256, 256, 0, stream>>>(c_mol, N_MOLS * HS);
    wln_final<<<blocks, 128, 0, stream>>>(hA0, hB, atom_graph, bond_graph, mol_ids,
                                          W0T, W1T, W2T, c_mol, c_atom);
}
